// MyModel_12738873000467
// MI455X (gfx1250) — matrix operand order repaired and checked by reading, not compiled, not run on hardware
//
#include <hip/hip_runtime.h>

typedef __attribute__((ext_vector_type(16))) _Float16 v16h;
typedef __attribute__((ext_vector_type(8)))  _Float16 v8h;
typedef __attribute__((ext_vector_type(8)))  float    v8f;

#define M_TILE    128
#define THREADS   256
#define NFEAT     1024
#define KIN       8

// ---- LDS layout (dynamic shared; total 309,760 B < 320 KB WGP LDS) ----
#define H1_STRIDE 1032                        // halfs per h1 row (padded: 516 dw -> 4-bank skew)
#define H1_BYTES  (M_TILE * H1_STRIDE * 2)    // 264192
#define W1F_STRIDE 9                          // padded (9 invertible mod 64 -> conflict-free)
#define W1F_OFF   H1_BYTES
#define W1F_BYTES (NFEAT * W1F_STRIDE * 4)    // 36864
#define B1F_OFF   (W1F_OFF + W1F_BYTES)
#define B1F_BYTES (NFEAT * 4)
#define U_OFF     (B1F_OFF + B1F_BYTES)
#define U_BYTES   (M_TILE * KIN * 4)
#define OACC_OFF  (U_OFF + U_BYTES)
#define SMEM_BYTES (OACC_OFF + M_TILE * 4)    // 309760

// rsqrt(1 + 1e-5) folded BN scale
#define BN_INV 0.99999500003749968f

// inter1d over DP_KP / PID_KP (inter2d results in the reference are discarded;
// PID_KI is constant so Ki = 0.1503906 * kp_rate exactly)
__device__ __forceinline__ float interp_kp(float d) {
    const float xs[6] = {-10.f, -2.5f, -1.f, 1.f, 2.5f, 10.f};
    const float ys[6] = {0.5f, 0.3417968f, 0.3417968f, 0.3417968f, 0.3417968f, 0.5f};
    int cnt = 0;
#pragma unroll
    for (int i = 0; i < 6; ++i) cnt += (xs[i] < d) ? 1 : 0;
    int idx = cnt - 1;
    idx = idx < 0 ? 0 : (idx > 4 ? 4 : idx);
    return ys[idx] + (d - xs[idx]) * (ys[idx + 1] - ys[idx]) / (xs[idx + 1] - xs[idx]);
}

// ---- W2 f32 -> f16 pre-conversion (used when workspace is large enough) ----
__global__ void convert_w2_f16(const float* __restrict__ W2, _Float16* __restrict__ W2h) {
    size_t i = ((size_t)blockIdx.x * THREADS + threadIdx.x) * 4;
#pragma unroll
    for (int k = 0; k < 4; ++k) W2h[i + k] = (_Float16)W2[i + k];
}

// ---- Fused PID-features + 3-layer MLP, layer-2 on v_wmma_f32_16x16x32_f16 ----
template <bool USE_WS>
__global__ __launch_bounds__(THREADS, 1) void hvac_mlp_kernel(
    const float* __restrict__ x,
    const float* __restrict__ W1, const float* __restrict__ b1,
    const float* __restrict__ g1, const float* __restrict__ be1,
    const float* __restrict__ W2f, const _Float16* __restrict__ W2h,
    const float* __restrict__ b2, const float* __restrict__ g2,
    const float* __restrict__ be2,
    const float* __restrict__ W3, const float* __restrict__ b3,
    float* __restrict__ out, int nrows)
{
    extern __shared__ char smem[];
    _Float16* h1     = (_Float16*)smem;                 // [M_TILE][H1_STRIDE] f16
    float*    w1f    = (float*)(smem + W1F_OFF);        // [NFEAT][9] folded W1
    float*    b1f    = (float*)(smem + B1F_OFF);        // [NFEAT]
    float*    u      = (float*)(smem + U_OFF);          // [M_TILE][8]
    float*    outAcc = (float*)(smem + OACC_OFF);       // [M_TILE]

    const int tid = threadIdx.x;
    const int rowBase = blockIdx.x * M_TILE;

    // ---------- Phase 0: fold BN1 into W1/b1 (in LDS) ----------
    for (int j = tid; j < NFEAT; j += THREADS) {
        float s = g1[j] * BN_INV;
#pragma unroll
        for (int k = 0; k < KIN; ++k) w1f[j * W1F_STRIDE + k] = W1[j * KIN + k] * s;
        b1f[j] = fmaf(b1[j], s, be1[j]);
    }

    // ---------- Phase 0b: per-row PID features ----------
    if (tid < M_TILE) {
        const int row = rowBase + tid;
        float lo_p = 0.f, hi_p = 0.f, aim_lo = 0.f, aim_hi = 0.f, kp_rate = 0.f;
        if (row < nrows) {
            const float* xr = x + (size_t)row * 16;
            lo_p = xr[8]; hi_p = xr[9]; aim_lo = xr[10]; aim_hi = xr[11]; kp_rate = xr[15];
        }
        const float diff_hi = hi_p - aim_hi;
        const float d = diff_hi * 0.001f;
        float* ur = u + tid * KIN;
        ur[0] = interp_kp(d) * kp_rate;      // Kp
        ur[1] = 0.1503906f * kp_rate;        // Ki (constant table)
        ur[2] = 0.0097656f * kp_rate;        // Kd
        ur[3] = diff_hi;
        ur[4] = lo_p;
        ur[5] = hi_p;
        ur[6] = aim_lo;
        ur[7] = aim_hi;
        outAcc[tid] = 0.f;
    }
    __syncthreads();

    // ---------- Phase 1: layer 1 (K=8, VALU) -> h1 f16 tile in LDS ----------
    for (int idx = tid; idx < M_TILE * NFEAT; idx += THREADS) {
        const int j = idx & (NFEAT - 1);
        const int m = idx >> 10;
        const float* ur = u + m * KIN;
        const float* wr = w1f + j * W1F_STRIDE;
        float acc = b1f[j];
#pragma unroll
        for (int k = 0; k < KIN; ++k) acc = fmaf(ur[k], wr[k], acc);
        h1[m * H1_STRIDE + j] = (_Float16)tanhf(acc);
    }
    __syncthreads();

    // ---------- Phase 2: layer 2 (WMMA f16, f32 acc) fused with layer 3 ----------
    const int lane  = tid & 31;
    const int wave  = tid >> 5;
    const int lhalf = lane >> 4;     // 0 / 1
    const int lmod  = lane & 15;
    const int kb    = lhalf * 16;    // B fragment K-chunk base
    const int aOff  = lhalf * 16;    // A fragment K-chunk offset

#pragma unroll 1
    for (int ntl = 0; ntl < 8; ++ntl) {
        const int n = wave * 128 + ntl * 16 + lmod;   // per-lane output column

        v8f acc[8];
#pragma unroll
        for (int r = 0; r < 8; ++r) acc[r] = {};

        const size_t brow = (size_t)n * NFEAT;
        for (int kk = 0; kk < NFEAT; kk += 32) {
            // B fragment: 32x16 (KxN), lane holds column n, 16 contiguous K
            v16h bv;
            if constexpr (USE_WS) {
                bv = *(const v16h*)(W2h + brow + kk + kb);
            } else {
                const float* bp = W2f + brow + kk + kb;
                v8f f0 = *(const v8f*)(bp);
                v8f f1 = *(const v8f*)(bp + 8);
#pragma unroll
                for (int i = 0; i < 8; ++i) { bv[i] = (_Float16)f0[i]; bv[i + 8] = (_Float16)f1[i]; }
            }
#pragma unroll
            for (int rt = 0; rt < 8; ++rt) {
                // A fragment: 16x32 (MxK): lane -> row, two 8-half chunks
                const _Float16* ap = h1 + (rt * 16 + lmod) * H1_STRIDE + kk;
                v8h alo = *(const v8h*)(ap + aOff);
                v8h ahi = *(const v8h*)(ap + 8 + aOff);
                v16h av = __builtin_shufflevector(alo, ahi,
                    0, 1, 2, 3, 4, 5, 6, 7, 8, 9, 10, 11, 12, 13, 14, 15);
                acc[rt] = __builtin_amdgcn_wmma_f32_16x16x32_f16(
                    false, av, false, bv, (short)0, acc[rt], false, false);
            }
        }

        // Epilogue: BN2 affine + tanh + dot with w3, reduce across N lanes
        const float a2  = g2[n] * BN_INV;
        const float c2  = fmaf(b2[n], a2, be2[n]);
        const float w3v = W3[n];
#pragma unroll
        for (int rt = 0; rt < 8; ++rt) {
#pragma unroll
            for (int v = 0; v < 8; ++v) {
                float hv = tanhf(fmaf(acc[rt][v], a2, c2));
                float contrib = hv * w3v;
                contrib += __shfl_xor(contrib, 1, 32);
                contrib += __shfl_xor(contrib, 2, 32);
                contrib += __shfl_xor(contrib, 4, 32);
                contrib += __shfl_xor(contrib, 8, 32);
                if (lmod == 0) {
                    const int m = rt * 16 + v + lhalf * 8;  // C/D layout: lanes>=16 -> M=v+8
                    atomicAdd(&outAcc[m], contrib);
                }
            }
        }
    }
    __syncthreads();

    // ---------- Final: out = acc + b3 ----------
    if (tid < M_TILE) {
        const int row = rowBase + tid;
        if (row < nrows) out[row] = outAcc[tid] + b3[0];
    }
}

extern "C" void kernel_launch(void* const* d_in, const int* in_sizes, int n_in,
                              void* d_out, int out_size, void* d_ws, size_t ws_size,
                              hipStream_t stream) {
    const float* x   = (const float*)d_in[0];
    const float* W1  = (const float*)d_in[1];
    const float* b1  = (const float*)d_in[2];
    const float* g1  = (const float*)d_in[3];
    const float* be1 = (const float*)d_in[4];
    const float* W2  = (const float*)d_in[5];
    const float* b2  = (const float*)d_in[6];
    const float* g2  = (const float*)d_in[7];
    const float* be2 = (const float*)d_in[8];
    const float* W3  = (const float*)d_in[9];
    const float* b3  = (const float*)d_in[10];
    float* out = (float*)d_out;

    const int nrows  = in_sizes[0] / 16;
    const int blocks = (nrows + M_TILE - 1) / M_TILE;

    const size_t w2h_bytes = (size_t)NFEAT * NFEAT * sizeof(_Float16);
    const bool useWs = (d_ws != nullptr) && (ws_size >= w2h_bytes);

    if (useWs) {
        _Float16* W2h = (_Float16*)d_ws;
        convert_w2_f16<<<(NFEAT * NFEAT) / (THREADS * 4), THREADS, 0, stream>>>(W2, W2h);
        (void)hipFuncSetAttribute((const void*)&hvac_mlp_kernel<true>,
                                  hipFuncAttributeMaxDynamicSharedMemorySize, SMEM_BYTES);
        hvac_mlp_kernel<true><<<blocks, THREADS, SMEM_BYTES, stream>>>(
            x, W1, b1, g1, be1, W2, W2h, b2, g2, be2, W3, b3, out, nrows);
    } else {
        (void)hipFuncSetAttribute((const void*)&hvac_mlp_kernel<false>,
                                  hipFuncAttributeMaxDynamicSharedMemorySize, SMEM_BYTES);
        hvac_mlp_kernel<false><<<blocks, THREADS, SMEM_BYTES, stream>>>(
            x, W1, b1, g1, be1, W2, (const _Float16*)nullptr, b2, g2, be2, W3, b3, out, nrows);
    }
}
